// MultiHeadAttention_39470749450981
// MI455X (gfx1250) — compile-verified, matrix operand order checked by reading, not run on hardware
//
#include <hip/hip_runtime.h>
#include <hip/hip_bf16.h>

// MHA: N=2, LQ=LK=2048, D=1024, H=16, head_dim=64.
// bf16 WMMA (v_wmma_f32_16x16x32_bf16) projections + flash attention.
// Attention K/V tiles are double-buffered in LDS and staged by the CDNA5
// Tensor Data Mover (tensor_load_to_lds, TENSORcnt) so DMA of tile kb+1
// overlaps the WMMAs on tile kb. Workspace (>= 24 MB): bf16 Q|K|V [N][H][L][64].

#define N_B  2
#define LQ_  2048
#define LK_  2048
#define DM_  1024
#define NH_  16
#define HD_  64

typedef __attribute__((ext_vector_type(16))) __bf16 bf16x16;
typedef __attribute__((ext_vector_type(8)))  float  f32x8;
typedef __attribute__((ext_vector_type(4)))  unsigned int uint32x4;
typedef __attribute__((ext_vector_type(8)))  int int32x8;
typedef __attribute__((ext_vector_type(4)))  int int32x4;

union BfFrag { bf16x16 v; unsigned short u[16]; };

__device__ __forceinline__ unsigned short f2bf(float f) {
  __hip_bfloat16 h = __float2bfloat16(f);
  unsigned short s;
  __builtin_memcpy(&s, &h, sizeof(s));
  return s;
}
__device__ __forceinline__ unsigned int pack2(float a, float b) {
  float2 t; t.x = a; t.y = b;
  __hip_bfloat162 h2 = __float22bfloat162_rn(t);
  unsigned int u;
  __builtin_memcpy(&u, &h2, sizeof(u));
  return u;
}
// 16-bit A-matrix 16x32 K index for element e (0..15) in half-wave hf (0/1).
__device__ __forceinline__ int kmap(int e, int hf) {
  return ((e & 8) << 1) + (hf << 3) + (e & 7);
}

// ---------------------------------------------------------------------------
// Tensor Data Mover: DMA a 2D bf16 tile (rows x 64) of a [tensorRows][64]
// row-major tensor from global into LDS (contiguous, stride 64 elements).
// D# layout per CDNA5 ISA.
// ---------------------------------------------------------------------------
#if __has_builtin(__builtin_amdgcn_tensor_load_to_lds) && \
    __has_builtin(__builtin_amdgcn_s_wait_tensorcnt)
#define USE_TDM 1
#warning "CDNA5: TDM path ENABLED (tensor_load_to_lds + s_wait_tensorcnt)"
__device__ __forceinline__ void tdm_load_2d(const void* gptr, unsigned ldsOff,
                                            unsigned rows, unsigned tensorRows) {
  const unsigned long long ga = (unsigned long long)gptr;
  uint32x4 g0;
  g0[0] = 1u;                                        // count=1, user descriptor
  g0[1] = ldsOff;                                    // lds_addr (bytes)
  g0[2] = (unsigned)(ga & 0xffffffffu);              // global_addr[31:0]
  g0[3] = (unsigned)((ga >> 32) & 0x1ffffffu)        // global_addr[56:32]
        | (2u << 30);                                // type = 2 ("image")
  int32x8 g1;
  g1[0] = 0x00010000;                 // wg_mask=0, data_size=1 (2 bytes)
  g1[1] = (int)(64u << 16);           // tensor_dim0 = 64 (low 16 of bits 79:48)
  g1[2] = (int)(tensorRows << 16);    // dim0 hi = 0 | tensor_dim1 low 16
  g1[3] = (int)(64u << 16);           // tensor_dim1 hi = 0 | tile_dim0 = 64
  g1[4] = (int)rows;                  // tile_dim1 = rows, tile_dim2 = 0
  g1[5] = 64;                         // tensor_dim0_stride = 64 (low 32)
  g1[6] = 0;
  g1[7] = 0;
  const int32x4 z4 = {0, 0, 0, 0};
#if __clang_major__ >= 23
  const int32x8 z8 = {0, 0, 0, 0, 0, 0, 0, 0};
  __builtin_amdgcn_tensor_load_to_lds(g0, g1, z4, z4, z8, 0);
#else
  __builtin_amdgcn_tensor_load_to_lds(g0, g1, z4, z4, 0);
#endif
}
#else
#define USE_TDM 0
#warning "CDNA5: TDM builtins UNAVAILABLE - using vector-copy fallback"
#endif

// ---------------------------------------------------------------------------
// Kernel 1: Y = X @ W^T, f32 in -> bf16 out, split into heads.
// grid: (LQ/64, DM/64, 3*N)  block: 128 (4 waves)
// ---------------------------------------------------------------------------
__global__ __launch_bounds__(128)
void proj_qkv_kernel(const float* __restrict__ query, const float* __restrict__ keyin,
                     const float* __restrict__ Wq, const float* __restrict__ Wk,
                     const float* __restrict__ Wv, unsigned short* __restrict__ wsout) {
  const int XS = 36;                       // padded LDS stride (ushorts)
  __shared__ unsigned short ldsX[64 * 36];
  __shared__ unsigned short ldsW[64 * 36];

  const int tid  = threadIdx.x;
  const int lane = tid & 31, wave = tid >> 5;
  const int hf   = lane >> 4, ln = lane & 15;
  const int rowBase = blockIdx.x * 64;
  const int colBase = blockIdx.y * 64;
  const int nb = blockIdx.z / 3, pj = blockIdx.z % 3;

  const float* X = (pj == 0) ? (query + (size_t)nb * LQ_ * DM_)
                             : (keyin + (size_t)nb * LK_ * DM_);
  const float* W = (pj == 0) ? Wq : ((pj == 1) ? Wk : Wv);
  unsigned short* Y = wsout + (size_t)pj * ((size_t)N_B * NH_ * LQ_ * HD_);

  f32x8 acc[4];
  for (int g = 0; g < 4; ++g)
    for (int v = 0; v < 8; ++v) acc[g][v] = 0.0f;

  for (int kb = 0; kb < DM_ / 32; ++kb) {
    #pragma unroll
    for (int g = 0; g < 4; ++g) {
      const int c = tid + g * 128;
      const int r = c >> 3, cc = (c & 7) << 2;
      const float4 xv = *(const float4*)(X + (size_t)(rowBase + r) * DM_ + kb * 32 + cc);
      *(uint2*)&ldsX[r * XS + cc] = make_uint2(pack2(xv.x, xv.y), pack2(xv.z, xv.w));
      const float4 wv = *(const float4*)(W + (size_t)(colBase + r) * DM_ + kb * 32 + cc);
      *(uint2*)&ldsW[r * XS + cc] = make_uint2(pack2(wv.x, wv.y), pack2(wv.z, wv.w));
    }
    __syncthreads();

    BfFrag a;
    #pragma unroll
    for (int e = 0; e < 16; ++e)
      a.u[e] = ldsX[(wave * 16 + ln) * XS + kmap(e, hf)];
    #pragma unroll
    for (int g = 0; g < 4; ++g) {
      BfFrag b;
      #pragma unroll
      for (int e = 0; e < 16; ++e)
        b.u[e] = ldsW[(g * 16 + ln) * XS + kmap(e, hf)];
      acc[g] = __builtin_amdgcn_wmma_f32_16x16x32_bf16(
          false, a.v, false, b.v, (short)0, acc[g], false, false);
    }
    __syncthreads();
  }

  #pragma unroll
  for (int g = 0; g < 4; ++g)
    #pragma unroll
    for (int v = 0; v < 8; ++v) {
      const int row = rowBase + wave * 16 + v + (hf << 3);
      const int o   = colBase + g * 16 + ln;
      const int h   = o >> 6, dc = o & 63;
      Y[(((size_t)nb * NH_ + h) * LQ_ + row) * HD_ + dc] = f2bf(acc[g][v]);
    }
}

// ---------------------------------------------------------------------------
// Kernel 2: flash attention, double-buffered TDM K/V staging, online softmax.
// grid: (LQ/64, H, N)  block: 128 (4 waves; wave owns 16 query rows)
// ---------------------------------------------------------------------------
__global__ __launch_bounds__(128)
void attn_kernel(const unsigned short* __restrict__ wsQ,
                 const unsigned short* __restrict__ wsK,
                 const unsigned short* __restrict__ wsV,
                 const unsigned char* __restrict__ pad,
                 float* __restrict__ out) {
  const int PS = 36;
  __shared__ unsigned short ldsQ[64 * 64];
  __shared__ unsigned short ldsK[2][32 * 64];   // double buffer
  __shared__ unsigned short ldsV[2][32 * 64];
  __shared__ unsigned short ldsP[4][16 * 36];
  __shared__ int s_masked;

  const int tid  = threadIdx.x;
  const int lane = tid & 31, wave = tid >> 5;
  const int hf   = lane >> 4, ln = lane & 15;
  const int qBase = blockIdx.x * 64;
  const int h = blockIdx.y, nb = blockIdx.z;

  const unsigned short* Qh = wsQ + (((size_t)nb * NH_ + h) * LQ_) * HD_;
  const unsigned short* Kh = wsK + (((size_t)nb * NH_ + h) * LK_) * HD_;
  const unsigned short* Vh = wsV + (((size_t)nb * NH_ + h) * LK_) * HD_;
  const unsigned char* padn = pad + (size_t)nb * LK_;

  const int nkb = (qBase + 64) / 32;            // causal clamp on key blocks

  if (tid == 0) s_masked = 0;
  __syncthreads();

#if USE_TDM
  const unsigned offQ  = (unsigned)(size_t)(const void*)&ldsQ[0];
  const unsigned offK0 = (unsigned)(size_t)(const void*)&ldsK[0][0];
  const unsigned offK1 = (unsigned)(size_t)(const void*)&ldsK[1][0];
  const unsigned offV0 = (unsigned)(size_t)(const void*)&ldsV[0][0];
  const unsigned offV1 = (unsigned)(size_t)(const void*)&ldsV[1][0];
  if (wave == 0) {
    tdm_load_2d(Qh + (size_t)qBase * HD_, offQ, 64, LQ_);   // Q tile
    tdm_load_2d(Kh, offK0, 32, LK_);                        // K tile 0
    tdm_load_2d(Vh, offV0, 32, LK_);                        // V tile 0
  }
#else
  #pragma unroll
  for (int g = 0; g < 8; ++g) {
    const int c = tid + g * 128;
    const int r = c >> 4, cc = (c & 15) << 2;
    *(uint2*)&ldsQ[r * 64 + cc] = *(const uint2*)(Qh + (size_t)(qBase + r) * HD_ + cc);
  }
  #pragma unroll
  for (int g = 0; g < 4; ++g) {
    const int c = tid + g * 128;
    const int r = c >> 4, cc = (c & 15) << 2;
    *(uint2*)&ldsK[0][r * 64 + cc] = *(const uint2*)(Kh + (size_t)r * HD_ + cc);
    *(uint2*)&ldsV[0][r * 64 + cc] = *(const uint2*)(Vh + (size_t)r * HD_ + cc);
  }
#endif

  // count padded keys for the 1/sqrt(#valid) scaling (overlaps staging)
  int cnt = 0;
  for (int i = tid; i < LK_; i += 128) cnt += padn[i] ? 1 : 0;
  atomicAdd(&s_masked, cnt);

#if USE_TDM
  __builtin_amdgcn_s_wait_tensorcnt(0);         // Q + tile 0 landed
#endif
  __syncthreads();

  const float rscale = rsqrtf((float)(LK_ - s_masked));

  BfFrag aq0, aq1;
  #pragma unroll
  for (int e = 0; e < 16; ++e) {
    aq0.u[e] = ldsQ[(wave * 16 + ln) * 64 +      kmap(e, hf)];
    aq1.u[e] = ldsQ[(wave * 16 + ln) * 64 + 32 + kmap(e, hf)];
  }

  float rmax[8], rsum[8];
  f32x8 o[4];
  for (int v = 0; v < 8; ++v) { rmax[v] = -1e30f; rsum[v] = 0.0f; }
  for (int g = 0; g < 4; ++g)
    for (int v = 0; v < 8; ++v) o[g][v] = 0.0f;

  for (int kb = 0; kb < nkb; ++kb) {
    const int kBase = kb * 32;
    const int cur = kb & 1, nxt = cur ^ 1;

    // stage tile kb+1 into the other buffer; overlap with compute on tile kb
    if (kb + 1 < nkb) {
      const size_t goff = (size_t)(kBase + 32) * HD_;
#if USE_TDM
      if (wave == 0) {
        tdm_load_2d(Kh + goff, nxt ? offK1 : offK0, 32, LK_);
        tdm_load_2d(Vh + goff, nxt ? offV1 : offV0, 32, LK_);
      }
      __builtin_amdgcn_s_wait_tensorcnt(2);     // tile kb done; kb+1 in flight
#else
      #pragma unroll
      for (int g = 0; g < 4; ++g) {
        const int c = tid + g * 128;
        const int r = c >> 4, cc = (c & 15) << 2;
        *(uint2*)&ldsK[nxt][r * 64 + cc] = *(const uint2*)(Kh + goff + (size_t)r * HD_ + cc);
        *(uint2*)&ldsV[nxt][r * 64 + cc] = *(const uint2*)(Vh + goff + (size_t)r * HD_ + cc);
      }
#endif
      // warm GL2 two tiles ahead
      __builtin_prefetch(Kh + (size_t)(kBase + 64) * HD_ + tid * 16, 0, 1);
      __builtin_prefetch(Vh + (size_t)(kBase + 64) * HD_ + tid * 16, 0, 1);
    }
#if USE_TDM
    else {
      __builtin_amdgcn_s_wait_tensorcnt(0);     // last tile: drain
    }
#endif
    __syncthreads();

    // S = Q K^T  (16 rows x 32 keys per wave; contraction over d=64)
    f32x8 s0, s1;
    for (int v = 0; v < 8; ++v) { s0[v] = 0.0f; s1[v] = 0.0f; }
    {
      BfFrag b;
      #pragma unroll
      for (int e = 0; e < 16; ++e) b.u[e] = ldsK[cur][ln * 64 + kmap(e, hf)];
      s0 = __builtin_amdgcn_wmma_f32_16x16x32_bf16(false, aq0.v, false, b.v, (short)0, s0, false, false);
      #pragma unroll
      for (int e = 0; e < 16; ++e) b.u[e] = ldsK[cur][ln * 64 + 32 + kmap(e, hf)];
      s0 = __builtin_amdgcn_wmma_f32_16x16x32_bf16(false, aq1.v, false, b.v, (short)0, s0, false, false);
      #pragma unroll
      for (int e = 0; e < 16; ++e) b.u[e] = ldsK[cur][(16 + ln) * 64 + kmap(e, hf)];
      s1 = __builtin_amdgcn_wmma_f32_16x16x32_bf16(false, aq0.v, false, b.v, (short)0, s1, false, false);
      #pragma unroll
      for (int e = 0; e < 16; ++e) b.u[e] = ldsK[cur][(16 + ln) * 64 + 32 + kmap(e, hf)];
      s1 = __builtin_amdgcn_wmma_f32_16x16x32_bf16(false, aq1.v, false, b.v, (short)0, s1, false, false);
    }

    // scale + causal/padding mask + online softmax
    const int  kq0 = kBase + ln, kq1 = kBase + 16 + ln;
    const bool pm0 = padn[kq0] != 0, pm1 = padn[kq1] != 0;
    float alpha[8];
    #pragma unroll
    for (int v = 0; v < 8; ++v) {
      const int qrow = qBase + wave * 16 + v + (hf << 3);
      const float x0 = (kq0 > qrow || pm0) ? -1e30f : s0[v] * rscale;
      const float x1 = (kq1 > qrow || pm1) ? -1e30f : s1[v] * rscale;
      float t = fmaxf(x0, x1);
      t = fmaxf(t, __shfl_xor(t, 1, 32));
      t = fmaxf(t, __shfl_xor(t, 2, 32));
      t = fmaxf(t, __shfl_xor(t, 4, 32));
      t = fmaxf(t, __shfl_xor(t, 8, 32));
      const float mnew = fmaxf(rmax[v], t);
      alpha[v] = __expf(rmax[v] - mnew);
      rmax[v]  = mnew;
      const float p0 = __expf(x0 - mnew);
      const float p1 = __expf(x1 - mnew);
      float ps = p0 + p1;
      ps += __shfl_xor(ps, 1, 32);
      ps += __shfl_xor(ps, 2, 32);
      ps += __shfl_xor(ps, 4, 32);
      ps += __shfl_xor(ps, 8, 32);
      rsum[v] = rsum[v] * alpha[v] + ps;
      const int m = v + (hf << 3);
      ldsP[wave][m * PS + ln]      = f2bf(p0);
      ldsP[wave][m * PS + 16 + ln] = f2bf(p1);
    }
    #pragma unroll
    for (int g = 0; g < 4; ++g)
      #pragma unroll
      for (int v = 0; v < 8; ++v) o[g][v] *= alpha[v];

    // fence the intra-wave LDS P round-trip (DS ops are in-order per wave)
    asm volatile("s_wait_dscnt 0" ::: "memory");

    // O += P @ V
    BfFrag ap;
    #pragma unroll
    for (int e = 0; e < 16; ++e)
      ap.u[e] = ldsP[wave][ln * PS + kmap(e, hf)];
    #pragma unroll
    for (int g = 0; g < 4; ++g) {
      BfFrag bv;
      #pragma unroll
      for (int e = 0; e < 16; ++e)
        bv.u[e] = ldsV[cur][kmap(e, hf) * 64 + g * 16 + ln];
      o[g] = __builtin_amdgcn_wmma_f32_16x16x32_bf16(
          false, ap.v, false, bv.v, (short)0, o[g], false, false);
    }
    __syncthreads();   // all waves done with buffer 'cur' before it is re-DMAed
  }

  #pragma unroll
  for (int v = 0; v < 8; ++v) {
    const int qrow = qBase + wave * 16 + v + (hf << 3);
    const float inv = (rsum[v] > 0.0f) ? (1.0f / rsum[v]) : 0.0f;
    #pragma unroll
    for (int g = 0; g < 4; ++g)
      out[((size_t)nb * LQ_ + qrow) * DM_ + h * HD_ + g * 16 + ln] = o[g][v] * inv;
  }
}

// ---------------------------------------------------------------------------
extern "C" void kernel_launch(void* const* d_in, const int* in_sizes, int n_in,
                              void* d_out, int out_size, void* d_ws, size_t ws_size,
                              hipStream_t stream) {
  const float* query = (const float*)d_in[0];
  const float* keyin = (const float*)d_in[1];
  const float* Wq    = (const float*)d_in[2];
  const float* Wk    = (const float*)d_in[3];
  const float* Wv    = (const float*)d_in[4];
  // d_in[5] = causal mask (recomputed arithmetically), d_in[6] = padding mask
  const unsigned char* pad = (const unsigned char*)d_in[6];
  // d_in[7] = n_heads (compile-time constant here)

  unsigned short* ws = (unsigned short*)d_ws;   // needs >= 24 MB
  const size_t perMat = (size_t)N_B * NH_ * LQ_ * HD_;

  proj_qkv_kernel<<<dim3(LQ_ / 64, DM_ / 64, 3 * N_B), 128, 0, stream>>>(
      query, keyin, Wq, Wk, Wv, ws);
  attn_kernel<<<dim3(LQ_ / 64, NH_, N_B), 128, 0, stream>>>(
      ws, ws + perMat, ws + 2 * perMat, pad, (float*)d_out);
}
